// Attention_23502061044268
// MI455X (gfx1250) — matrix operand order repaired and checked by reading, not compiled, not run on hardware
//
#include <hip/hip_runtime.h>
#include <math.h>

#define B_  32
#define N_  4096
#define D_  1024
#define H_  16
#define DK_ 64

typedef __attribute__((ext_vector_type(16))) __bf16 v16bf;
typedef __attribute__((ext_vector_type(8)))  __bf16 v8bf;
typedef __attribute__((ext_vector_type(8)))  float  v8f;

// ---------------------------------------------------------------------------
// Kernel 1: u[b,h,e] = (1/8) * sum_d qt[b,h,d] * wk[e, h*64+d]
//           qt[b,h,d] = sum_e query[b,e] * wq[e, h*64+d]
// ---------------------------------------------------------------------------
__global__ void k1_proj_u(const float* __restrict__ query,
                          const float* __restrict__ wq,
                          const float* __restrict__ wk,
                          __bf16* __restrict__ u) {
    __shared__ float qt[DK_];
    const int bh = blockIdx.x;
    const int b = bh >> 4, h = bh & 15;
    const int d = threadIdx.x;              // 0..63
    // q~ (one element per thread)
    float acc = 0.f;
    const float* qrow = query + (size_t)b * D_;
    const float* wqc  = wq + h * DK_ + d;   // column stride = 1024
    for (int e = 0; e < D_; ++e) acc += qrow[e] * wqc[(size_t)e * D_];
    qt[d] = acc;
    __syncthreads();
    // u: 16 e-values per thread
    for (int i = 0; i < 16; ++i) {
        const int e = d * 16 + i;
        const float* wkrow = wk + (size_t)e * D_ + h * DK_;
        float s = 0.f;
        #pragma unroll 8
        for (int dd = 0; dd < DK_; ++dd) s += qt[dd] * wkrow[dd];
        u[(size_t)bh * D_ + e] = (__bf16)(s * 0.125f);   // fold 1/sqrt(64)
    }
}

// ---------------------------------------------------------------------------
// Kernel 2: scores[b,h,n] = key[b,n,:] . u[b,h,:]   (WMMA bf16, f32 acc)
// Block: 256 thr = 8 waves, each wave owns a 16-row tile. Grid (N/128, B).
// WMMA tile: M = 16 seq rows, N = 16 heads, K over 1024 in steps of 32.
// ---------------------------------------------------------------------------
__global__ void k2_scores(const float* __restrict__ key,
                          const __bf16* __restrict__ u,
                          float* __restrict__ scores) {
    const int b    = blockIdx.y;
    const unsigned tid = threadIdx.x;
    const int wave = tid >> 5;
    const int lane = tid & 31;
    const int hf   = lane >> 4;     // lane half (ISA A/B layout)
    const int col  = lane & 15;     // A row-in-tile AND B column (head)
    const int n0   = blockIdx.x * 128 + wave * 16;

    const float*  keyrow = key + ((size_t)b * N_ + (n0 + col)) * D_;
    const __bf16* ub     = u   + ((size_t)b * H_ + col) * D_;

    v8f acc = {};
    for (int e0 = 0; e0 < D_; e0 += 32) {
        // A: key rows, fp32 -> bf16 in-register (traffic is fp32 regardless)
        v8f k0 = *(const v8f*)(keyrow + e0 + hf * 16);
        v8f k1 = *(const v8f*)(keyrow + e0 + hf * 16 + 8);
        v16bf a;
        #pragma unroll
        for (int j = 0; j < 8; ++j) { a[j] = (__bf16)k0[j]; a[8 + j] = (__bf16)k1[j]; }
        // B: u (already bf16, contiguous K=hf*16+j)
        v16bf bm = *(const v16bf*)(ub + e0 + hf * 16);
        acc = __builtin_amdgcn_wmma_f32_16x16x32_bf16(false, a, false, bm,
                                                      (short)0, acc, false, false);
    }
    // C/D: element i -> n = n0 + 8*hf + i (contiguous), head = col
    float* sp = scores + ((size_t)b * H_ + col) * N_ + n0 + hf * 8;
    *(v8f*)sp = acc;
}

// ---------------------------------------------------------------------------
// Kernel 3: per-(b,h) softmax; store unnormalized p = exp(s-max) as bf16
//           and the partition sum (normalization deferred to kernel 5).
// ---------------------------------------------------------------------------
__global__ void k3_softmax(const float* __restrict__ scores,
                           __bf16* __restrict__ p,
                           float* __restrict__ sums) {
    __shared__ float red[256];
    const int bh = blockIdx.x;
    const int t  = threadIdx.x;
    const float* s = scores + (size_t)bh * N_;
    float m = -1e30f;
    for (int i = t; i < N_; i += 256) m = fmaxf(m, s[i]);
    red[t] = m; __syncthreads();
    for (int off = 128; off > 0; off >>= 1) {
        if (t < off) red[t] = fmaxf(red[t], red[t + off]);
        __syncthreads();
    }
    m = red[0]; __syncthreads();
    float sum = 0.f;
    __bf16* pr = p + (size_t)bh * N_;
    for (int i = t; i < N_; i += 256) {
        float e = __expf(s[i] - m);
        sum += e;
        pr[i] = (__bf16)e;
    }
    red[t] = sum; __syncthreads();
    for (int off = 128; off > 0; off >>= 1) {
        if (t < off) red[t] += red[t + off];
        __syncthreads();
    }
    if (t == 0) sums[bh] = red[0];
}

// ---------------------------------------------------------------------------
// Kernel 4: w[z][b,h,e] = sum_{n in quarter z} p[b,h,n] * value[b,n,e]
// WMMA tile: M = 16 heads, N = 16-wide e-tile, K over 1024 (quarter) by 32.
// Grid (D/128, B, 4): 4-way K split into separate partials (deterministic).
// ---------------------------------------------------------------------------
__global__ void k4_pv(const float* __restrict__ value,
                      const __bf16* __restrict__ p,
                      float* __restrict__ wpart) {
    const int b    = blockIdx.y;
    const int z    = blockIdx.z;          // K quarter
    const unsigned tid = threadIdx.x;
    const int wave = tid >> 5;
    const int lane = tid & 31;
    const int hf   = lane >> 4;
    const int col  = lane & 15;           // A row (head) AND B column (e)
    const int e    = blockIdx.x * 128 + wave * 16 + col;

    const __bf16* prow = p + ((size_t)b * H_ + col) * N_;
    const float*  vb   = value + (size_t)b * N_ * D_ + e;

    const int nbeg = z * (N_ / 4);
    v8f acc = {};
    for (int n0 = nbeg; n0 < nbeg + N_ / 4; n0 += 32) {
        v8bf p0 = *(const v8bf*)(prow + n0 + hf * 16);
        v8bf p1 = *(const v8bf*)(prow + n0 + hf * 16 + 8);
        v16bf a;
        #pragma unroll
        for (int j = 0; j < 8; ++j) { a[j] = p0[j]; a[8 + j] = p1[j]; }
        // B: value rows n0+hf*16+j at fixed column e, fp32 -> bf16
        const float* vrow = vb + (size_t)(n0 + hf * 16) * D_;
        v16bf bm;
        #pragma unroll
        for (int j = 0; j < 16; ++j) bm[j] = (__bf16)vrow[(size_t)j * D_];
        acc = __builtin_amdgcn_wmma_f32_16x16x32_bf16(false, a, false, bm,
                                                      (short)0, acc, false, false);
    }
    // C/D: element i -> head = i + 8*hf, column e
    #pragma unroll
    for (int i = 0; i < 8; ++i) {
        const int h = i + 8 * hf;
        wpart[((size_t)z * B_ * H_ + (size_t)b * H_ + h) * D_ + e] = acc[i];
    }
}

// ---------------------------------------------------------------------------
// Kernel 5: out[b, h*64+d] = (sum_e (sum_z wpart) * wv[e, h*64+d]) / sums[bh]
// ---------------------------------------------------------------------------
__global__ void k5_out(const float* __restrict__ wpart,
                       const float* __restrict__ wv,
                       const float* __restrict__ sums,
                       float* __restrict__ out) {
    __shared__ float wrow[D_];
    const int bh = blockIdx.x;
    const int b = bh >> 4, h = bh & 15;
    const int d = threadIdx.x;            // 0..63
    for (int i = d; i < D_; i += 64) {
        float s = 0.f;
        #pragma unroll
        for (int z = 0; z < 4; ++z)
            s += wpart[((size_t)z * B_ * H_ + bh) * D_ + i];
        wrow[i] = s;
    }
    __syncthreads();
    float acc = 0.f;
    const float* wvc = wv + h * DK_ + d;  // column stride = 1024
    for (int e = 0; e < D_; ++e) acc += wrow[e] * wvc[(size_t)e * D_];
    out[(size_t)b * (H_ * DK_) + h * DK_ + d] = acc / sums[bh];
}

// ---------------------------------------------------------------------------
extern "C" void kernel_launch(void* const* d_in, const int* in_sizes, int n_in,
                              void* d_out, int out_size, void* d_ws, size_t ws_size,
                              hipStream_t stream) {
    const float* query = (const float*)d_in[0];
    const float* key   = (const float*)d_in[1];
    const float* value = (const float*)d_in[2];
    const float* wq    = (const float*)d_in[3];
    const float* wk    = (const float*)d_in[4];
    const float* wv    = (const float*)d_in[5];
    float* out = (float*)d_out;

    char* ws = (char*)d_ws;
    __bf16* u      = (__bf16*)(ws);                        // 1 MB
    float*  scores = (float*)(ws + (1u << 20));            // 8 MB
    float*  wpart  = (float*)(ws + (1u << 20));            // aliases scores (dead after k3)
    __bf16* p      = (__bf16*)(ws + (9u << 20));           // 4 MB
    float*  sums   = (float*)(ws + (13u << 20));           // 2 KB

    k1_proj_u <<<dim3(B_ * H_),        dim3(64),  0, stream>>>(query, wq, wk, u);
    k2_scores <<<dim3(N_ / 128, B_),   dim3(256), 0, stream>>>(key, u, scores);
    k3_softmax<<<dim3(B_ * H_),        dim3(256), 0, stream>>>(scores, p, sums);
    k4_pv     <<<dim3(D_ / 128, B_, 4),dim3(256), 0, stream>>>(value, p, wpart);
    k5_out    <<<dim3(B_ * H_),        dim3(64),  0, stream>>>(wpart, wv, sums, out);
}
